// RAN_91199335563545
// MI455X (gfx1250) — matrix operand order repaired and checked by reading, not compiled, not run on hardware
//
#include <hip/hip_runtime.h>
#include <hip/hip_bf16.h>

typedef __bf16 bf16;
typedef __attribute__((ext_vector_type(16))) __bf16        v16bf;
typedef __attribute__((ext_vector_type(8)))  float         v8f;
typedef __attribute__((ext_vector_type(4)))  unsigned int  v4u;

#define T_STEPS 512
#define BATCH   64
#define FEAT    512
#define HID     1024
#define H3      3072
#define TBm     (T_STEPS * BATCH)   // 32768 rows for the input projection

union Frag16 { v4u u[2]; v16bf v; };

// A fragment (16x32 bf16, MxK), row-major source with leading dim ldk.
// ISA 7.12.2: lanes 0-15 -> M=lane, K {0..7,16..23}; lanes 16-31 -> K {8..15,24..31}.
__device__ __forceinline__ v16bf load_frag_a(const bf16* base, int row, int ldk, int k, int half) {
    Frag16 f;
    const bf16* p = base + (size_t)row * ldk + k + half * 8;
    f.u[0] = *(const v4u*)p;          // 8 elems
    f.u[1] = *(const v4u*)(p + 16);   // 8 elems, +8 dwords
    return f.v;
}

// B fragment (32x16 bf16, KxN) where B[k][n] = W[n][k] (W row-major, leading dim ldk).
// lanes 0-15: N=lane, K {0..7,16..23}; lanes 16-31: N=lane-16, K {8..15,24..31}.
__device__ __forceinline__ v16bf load_frag_b(const bf16* base, int row, int ldk, int k, int half) {
    Frag16 f;
    const bf16* p = base + (size_t)row * ldk + k + half * 8;
    f.u[0] = *(const v4u*)p;
    f.u[1] = *(const v4u*)(p + 16);
    return f.v;
}

// ---------------- fp32 -> bf16 convert ----------------
__global__ void f32_to_bf16_kernel(const float* __restrict__ src, bf16* __restrict__ dst, int n) {
    int i = blockIdx.x * blockDim.x + threadIdx.x;
    if (i < n) dst[i] = (bf16)src[i];
}

__global__ void zero_hidden_kernel(float* __restrict__ hf, bf16* __restrict__ hb, int n) {
    int i = blockIdx.x * blockDim.x + threadIdx.x;
    if (i < n) { hf[i] = 0.0f; hb[i] = (bf16)0.0f; }
}

// ---------------- input projection: gi[TB,3H] = X[TB,F] @ Wih^T + b_ih ----------------
// 8 waves/WG (2x4) -> 128x128 WG tile; wave = 64x32 (4 Msub x 2 Nsub) = 8 WMMAs/k-step.
// Fragment loads double-buffered so k+32 loads overlap the k WMMAs.
__global__ __launch_bounds__(256) void ran_gemm_ih(const bf16* __restrict__ X,
                                                   const bf16* __restrict__ W,
                                                   const float* __restrict__ bias,
                                                   float* __restrict__ gi)
{
    const int lane = threadIdx.x & 31;
    const int wave = threadIdx.x >> 5;
    const int half = lane >> 4;
    const int l16  = lane & 15;
    const int tileM = blockIdx.x * 128 + (wave >> 2) * 64;
    const int tileN = blockIdx.y * 128 + (wave & 3) * 32;

    v8f acc[4][2] = {};
    v16bf a[2][4], b[2][2];

#pragma unroll
    for (int ms = 0; ms < 4; ++ms) a[0][ms] = load_frag_a(X, tileM + ms * 16 + l16, FEAT, 0, half);
#pragma unroll
    for (int ns = 0; ns < 2; ++ns) b[0][ns] = load_frag_b(W, tileN + ns * 16 + l16, FEAT, 0, half);

    int buf = 0;
    for (int k = 0; k < FEAT; k += 32) {
        const int nk = k + 32;
        if (nk < FEAT) {   // prefetch next k-block into the other buffer
#pragma unroll
            for (int ms = 0; ms < 4; ++ms)
                a[buf ^ 1][ms] = load_frag_a(X, tileM + ms * 16 + l16, FEAT, nk, half);
#pragma unroll
            for (int ns = 0; ns < 2; ++ns)
                b[buf ^ 1][ns] = load_frag_b(W, tileN + ns * 16 + l16, FEAT, nk, half);
        }
#pragma unroll
        for (int ms = 0; ms < 4; ++ms)
#pragma unroll
            for (int ns = 0; ns < 2; ++ns)
                acc[ms][ns] = __builtin_amdgcn_wmma_f32_16x16x32_bf16(
                    false, a[buf][ms], false, b[buf][ns], (short)0, acc[ms][ns], false, false);
        buf ^= 1;
    }

#pragma unroll
    for (int ms = 0; ms < 4; ++ms)
#pragma unroll
        for (int ns = 0; ns < 2; ++ns) {
            const int   n  = tileN + ns * 16 + l16;
            const float bn = bias[n];
#pragma unroll
            for (int r = 0; r < 8; ++r) {
                const int m = tileM + ms * 16 + half * 8 + r;  // C layout: VGPR r -> M=r / r+8
                gi[(size_t)m * H3 + n] = acc[ms][ns][r] + bn;
            }
        }
}

// ---------------- one recurrent step (fused dual-gate GEMM + pointwise) ----------------
// grid = HID/128 WGs of 8 waves; each wave: 64x16 per gate (4 Msub x 2 gates).
// Hidden state (64x1024 bf16 = 128KB) DMA'd into LDS via CDNA5 async-to-LDS path.
__global__ __launch_bounds__(256) void ran_step(const bf16*  __restrict__ hprev_bf,
                                                const float* __restrict__ hprev_f,
                                                const bf16*  __restrict__ Whh,
                                                const float* __restrict__ gi_t,
                                                const float* __restrict__ bias_hh,
                                                float* __restrict__ hnext_f,
                                                bf16*  __restrict__ hnext_bf,
                                                float* __restrict__ out_t)
{
    extern __shared__ bf16 sh[];   // BATCH*HID bf16 = 128 KB (fits 320 KB/WGP)
    const int tid = threadIdx.x;

    {   // async DMA h(t) -> LDS: global_load_async_to_lds_b128, tracked by ASYNCcnt
        const v4u* src = (const v4u*)hprev_bf;
        const int  total = (BATCH * HID) / 8;   // v4u = 8 bf16 = 16B per lane per op
        for (int i = tid; i < total; i += 256) {
            unsigned           ldsOff = (unsigned)(uintptr_t)((v4u*)sh + i);   // addr[31:0] = LDS offset
            unsigned long long ga     = (unsigned long long)(uintptr_t)(src + i);
            asm volatile("global_load_async_to_lds_b128 %0, %1, off"
                         :: "v"(ldsOff), "v"(ga) : "memory");
        }
        asm volatile("s_wait_asynccnt 0x0" ::: "memory");
    }
    __syncthreads();

    const int lane  = tid & 31;
    const int wave  = tid >> 5;
    const int half  = lane >> 4;
    const int l16   = lane & 15;
    const int tileN = blockIdx.x * 128 + wave * 16;

    v8f accI[4] = {};
    v8f accF[4] = {};

    v16bf a[2][4], bI[2], bF[2];
#pragma unroll
    for (int ms = 0; ms < 4; ++ms) a[0][ms] = load_frag_a(sh, ms * 16 + l16, HID, 0, half);
    bI[0] = load_frag_b(Whh, tileN + l16, HID, 0, half);
    bF[0] = load_frag_b(Whh, HID + tileN + l16, HID, 0, half);

    int buf = 0;
    for (int k = 0; k < HID; k += 32) {
        const int nk = k + 32;
        if (nk < HID) {
#pragma unroll
            for (int ms = 0; ms < 4; ++ms)
                a[buf ^ 1][ms] = load_frag_a(sh, ms * 16 + l16, HID, nk, half);   // ds_load_b128
            bI[buf ^ 1] = load_frag_b(Whh, tileN + l16, HID, nk, half);           // w_hi rows
            bF[buf ^ 1] = load_frag_b(Whh, HID + tileN + l16, HID, nk, half);     // w_hf rows
        }
#pragma unroll
        for (int ms = 0; ms < 4; ++ms) {
            accI[ms] = __builtin_amdgcn_wmma_f32_16x16x32_bf16(
                false, a[buf][ms], false, bI[buf], (short)0, accI[ms], false, false);
            accF[ms] = __builtin_amdgcn_wmma_f32_16x16x32_bf16(
                false, a[buf][ms], false, bF[buf], (short)0, accF[ms], false, false);
        }
        buf ^= 1;
    }

    const int   n   = tileN + l16;
    const float bhi = bias_hh[n];
    const float bhf = bias_hh[HID + n];
#pragma unroll
    for (int ms = 0; ms < 4; ++ms)
#pragma unroll
        for (int r = 0; r < 8; ++r) {
            const int    b   = ms * 16 + half * 8 + r;        // batch row
            const size_t gib = (size_t)b * H3;
            const float  pre_i = accI[ms][r] + gi_t[gib + n]        + bhi;
            const float  pre_f = accF[ms][r] + gi_t[gib + HID + n]  + bhf;
            const float  ig    = 1.0f / (1.0f + __expf(-pre_i));
            const float  fg    = 1.0f / (1.0f + __expf(-pre_f));
            const float  i_n   = gi_t[gib + 2 * HID + n];
            const float  hold  = hprev_f[(size_t)b * HID + n];    // fp32 carry path
            const float  hnew  = ig * i_n + fg * hold;
            hnext_f [(size_t)b * HID + n] = hnew;
            hnext_bf[(size_t)b * HID + n] = (bf16)hnew;
            out_t   [(size_t)b * HID + n] = tanhf(hnew);
        }
}

extern "C" void kernel_launch(void* const* d_in, const int* in_sizes, int n_in,
                              void* d_out, int out_size, void* d_ws, size_t ws_size,
                              hipStream_t stream)
{
    (void)in_sizes; (void)n_in; (void)out_size; (void)ws_size;

    const float* input = (const float*)d_in[0];   // [512,64,512]
    const float* w_ih  = (const float*)d_in[1];   // [3072,512]
    const float* w_hh  = (const float*)d_in[2];   // [2048,1024]
    const float* b_ih  = (const float*)d_in[3];   // [3072]
    const float* b_hh  = (const float*)d_in[4];   // [2048]
    float*       out   = (float*)d_out;           // [512,64,1024]

    char* ws = (char*)d_ws;
    size_t off = 0;
    auto carve = [&](size_t bytes) -> void* {
        void* p = (void*)(ws + off);
        off += (bytes + 255) & ~(size_t)255;
        return p;
    };

    bf16*  Xbf   = (bf16*)  carve((size_t)TBm * FEAT * sizeof(bf16));       // 32 MB
    bf16*  Wihbf = (bf16*)  carve((size_t)H3 * FEAT * sizeof(bf16));        // 3 MB
    bf16*  Whhbf = (bf16*)  carve((size_t)2 * HID * HID * sizeof(bf16));    // 4 MB
    float* gi    = (float*) carve((size_t)TBm * H3 * sizeof(float));        // 384 MB
    float* hf0   = (float*) carve((size_t)BATCH * HID * sizeof(float));
    float* hf1   = (float*) carve((size_t)BATCH * HID * sizeof(float));
    bf16*  hb0   = (bf16*)  carve((size_t)BATCH * HID * sizeof(bf16));
    bf16*  hb1   = (bf16*)  carve((size_t)BATCH * HID * sizeof(bf16));

    const int cv = 256;
    f32_to_bf16_kernel<<<(TBm * FEAT + cv - 1) / cv, cv, 0, stream>>>(input, Xbf, TBm * FEAT);
    f32_to_bf16_kernel<<<(H3 * FEAT + cv - 1) / cv, cv, 0, stream>>>(w_ih, Wihbf, H3 * FEAT);
    f32_to_bf16_kernel<<<(2 * HID * HID + cv - 1) / cv, cv, 0, stream>>>(w_hh, Whhbf, 2 * HID * HID);
    zero_hidden_kernel<<<(BATCH * HID + cv - 1) / cv, cv, 0, stream>>>(hf0, hb0, BATCH * HID);

    dim3 g1(TBm / 128, H3 / 128);   // 256 x 24 workgroups
    ran_gemm_ih<<<g1, 256, 0, stream>>>(Xbf, Wihbf, b_ih, gi);

    float* hf[2] = { hf0, hf1 };
    bf16*  hb[2] = { hb0, hb1 };
    const size_t ldsBytes = (size_t)BATCH * HID * sizeof(bf16);  // 128 KB dynamic LDS
    for (int t = 0; t < T_STEPS; ++t) {
        const int p = t & 1, q = p ^ 1;
        ran_step<<<HID / 128, 256, ldsBytes, stream>>>(
            hb[p], hf[p], Whhbf,
            gi + (size_t)t * BATCH * H3, b_hh,
            hf[q], hb[q],
            out + (size_t)t * BATCH * HID);
    }
}
